// CrossLocal_31834297598462
// MI455X (gfx1250) — matrix operand order repaired and checked by reading, not compiled, not run on hardware
//
#include <hip/hip_runtime.h>
#include <math.h>

// ---------------------------------------------------------------------------
// CrossLocal attention for MI455X (gfx1250): fused flash-attention with
// bf16 WMMA (v_wmma_f32_16x16x32_bf16), f32 accumulation, DPP softmax
// reductions, packed-b32 LDS P-transpose, 64-key tiles (4 QK + 4 PV WMMAs
// per softmax pass).
// ---------------------------------------------------------------------------

typedef __bf16 bf16;
typedef __attribute__((ext_vector_type(16))) bf16  v16bf;
typedef __attribute__((ext_vector_type(8)))  bf16  v8bf;
typedef __attribute__((ext_vector_type(8)))  float v8f;

#define B_    4
#define C_    64
#define CI_   32
#define HM_   128
#define HC_   64
#define NC_   4096          // HC*HC
#define BN_EPS 1e-5f

static __device__ __forceinline__ v8f wmma_bf16(v16bf a, v16bf b, v8f c) {
  // D = A(16x32) * B(32x16) + C   (f32 accum)
  return __builtin_amdgcn_wmma_f32_16x16x32_bf16(
      /*neg_a=*/false, a, /*neg_b=*/false, b,
      /*c_mod=*/(short)0, c, /*reuse_a=*/false, /*reuse_b=*/false);
}

// ---- DPP-based 16-lane reductions (pure VALU, no LDS pipe) ----------------
template <int CTRL>
static __device__ __forceinline__ float dppf(float x) {
  return __builtin_bit_cast(
      float, __builtin_amdgcn_mov_dpp(__builtin_bit_cast(int, x),
                                      CTRL, 0xF, 0xF, true));
}
static __device__ __forceinline__ float red16_max(float x) {
  x = fmaxf(x, dppf<0xB1>(x));    // quad_perm [1,0,3,2]  (xor 1)
  x = fmaxf(x, dppf<0x4E>(x));    // quad_perm [2,3,0,1]  (xor 2)
  x = fmaxf(x, dppf<0x141>(x));   // row_half_mirror      (cross quads)
  x = fmaxf(x, dppf<0x140>(x));   // row_mirror           (cross octets)
  return x;
}
static __device__ __forceinline__ float red16_sum(float x) {
  x += dppf<0xB1>(x);
  x += dppf<0x4E>(x);
  x += dppf<0x141>(x);
  x += dppf<0x140>(x);
  return x;
}

static __device__ __forceinline__ unsigned pack_bf16(float a, float b) {
  unsigned short lo = __builtin_bit_cast(unsigned short, (bf16)a);
  unsigned short hi = __builtin_bit_cast(unsigned short, (bf16)b);
  return (unsigned)lo | ((unsigned)hi << 16);
}

// ---------------------------------------------------------------------------
// Kernel 1: Q = g(cross), K = theta(cross)  ->  bf16 [B][Nc][32] row-major
// ---------------------------------------------------------------------------
__global__ __launch_bounds__(256)
void prep_qk_kernel(const float* __restrict__ cross,
                    const float* __restrict__ g_w,  const float* __restrict__ g_b,
                    const float* __restrict__ th_w, const float* __restrict__ th_b,
                    bf16* __restrict__ Qb, bf16* __restrict__ Kb) {
  __shared__ float sgw[CI_ * C_], stw[CI_ * C_], sgb[CI_], stb[CI_];
  for (int t = threadIdx.x; t < CI_ * C_; t += 256) { sgw[t] = g_w[t]; stw[t] = th_w[t]; }
  if (threadIdx.x < CI_) { sgb[threadIdx.x] = g_b[threadIdx.x]; stb[threadIdx.x] = th_b[threadIdx.x]; }
  __syncthreads();

  const int gid = blockIdx.x * 256 + threadIdx.x;   // b*Nc + n
  const int b = gid >> 12;
  const int n = gid & (NC_ - 1);

  float x[C_];
  const float* cb = cross + (size_t)b * C_ * NC_ + n;
#pragma unroll
  for (int c = 0; c < C_; ++c) x[c] = cb[(size_t)c * NC_];

  bf16* qo = Qb + (size_t)gid * CI_;
  bf16* ko = Kb + (size_t)gid * CI_;
  for (int o = 0; o < CI_; ++o) {     // rolled: keeps code size sane
    float q = sgb[o], k = stb[o];
#pragma unroll
    for (int c = 0; c < C_; ++c) { q += sgw[o * C_ + c] * x[c]; k += stw[o * C_ + c] * x[c]; }
    qo[o] = (bf16)q;
    ko[o] = (bf16)k;
  }
}

// ---------------------------------------------------------------------------
// Kernel 2: V = phi(avgpool2x2(main))  ->  bf16 TRANSPOSED [B][32][Nc]
// (128->64 bilinear, align_corners=False, is exactly 2x2 average pooling)
// ---------------------------------------------------------------------------
__global__ __launch_bounds__(256)
void prep_v_kernel(const float* __restrict__ main_in,
                   const float* __restrict__ ph_w, const float* __restrict__ ph_b,
                   bf16* __restrict__ Vt) {
  __shared__ float spw[CI_ * C_], spb[CI_];
  for (int t = threadIdx.x; t < CI_ * C_; t += 256) spw[t] = ph_w[t];
  if (threadIdx.x < CI_) spb[threadIdx.x] = ph_b[threadIdx.x];
  __syncthreads();

  const int gid = blockIdx.x * 256 + threadIdx.x;   // b*Nc + n
  const int b = gid >> 12;
  const int n = gid & (NC_ - 1);
  const int i = n >> 6, j = n & 63;

  float x[C_];
  const float* mb = main_in + (size_t)b * C_ * HM_ * HM_;
#pragma unroll
  for (int c = 0; c < C_; ++c) {
    const float* p = mb + (size_t)c * HM_ * HM_ + (2 * i) * HM_ + 2 * j;
    x[c] = 0.25f * (p[0] + p[1] + p[HM_] + p[HM_ + 1]);
  }
  for (int d = 0; d < CI_; ++d) {     // rolled
    float v = spb[d];
#pragma unroll
    for (int c = 0; c < C_; ++c) v += spw[d * C_ + c] * x[c];
    Vt[((size_t)(b * CI_ + d)) * NC_ + n] = (bf16)v;
  }
}

// ---------------------------------------------------------------------------
// Kernel 3: flash attention.  8 waves / block, each wave owns 16 query rows.
// Per iteration: 64 keys -> 4 QK^T WMMAs, one DPP online-softmax pass,
// P transposed through packed LDS (2 tiles), 4 PV WMMAs.
// No cross-wave synchronization in the hot loop.
// ---------------------------------------------------------------------------
__global__ __launch_bounds__(256)
void flash_attn_kernel(const bf16* __restrict__ Qb, const bf16* __restrict__ Kb,
                       const bf16* __restrict__ Vt, float* __restrict__ O) {
  // P staging, two 16x32 bf16 tiles per wave, column-pair interleaved:
  // element (m, c) of tile t lives at u32 index t*256 + m*16 + (c & 15),
  // low 16 bits = c<16, high 16 bits = c>=16.
  __shared__ unsigned p_lds[8][512];

  const int lane = threadIdx.x & 31;
  const int wave = threadIdx.x >> 5;
  const int half = lane >> 4;          // which 16-lane half
  const int l16  = lane & 15;
  const int b    = blockIdx.y;
  const int q0   = blockIdx.x * 128 + wave * 16;

  // ---- Q tile as WMMA A-operand (lane l16 = row M; two 8-elem runs) ----
  const bf16* qrow = Qb + ((size_t)(b * NC_ + q0 + l16)) * CI_;
  v16bf qa;
  {
    v8bf lo = *(const v8bf*)(qrow + half * 16);       // dims half*16 .. +7
    v8bf hi = *(const v8bf*)(qrow + half * 16 + 8);   // dims half*16+8 .. +15
#pragma unroll
    for (int e = 0; e < 8; ++e) { qa[e] = lo[e]; qa[8 + e] = hi[e]; }
  }

  v8f acc_lo = {};  // D layout: lane -> dim 0..15,  vgpr v -> row v+8*half
  v8f acc_hi = {};  // lane -> dim 16..31
  float mrun[8], lrun[8];
#pragma unroll
  for (int v = 0; v < 8; ++v) { mrun[v] = -INFINITY; lrun[v] = 0.0f; }

  const bf16* Kbase = Kb + (size_t)b * NC_ * CI_;
  const bf16* Vrow0 = Vt + ((size_t)(b * CI_ + l16)) * NC_;        // dim l16
  const bf16* Vrow1 = Vt + ((size_t)(b * CI_ + 16 + l16)) * NC_;   // dim 16+l16
  const bf16* Kpref = Kbase + (size_t)lane * CI_;                  // prefetch rows
  const bf16* Vpref = Vt + ((size_t)(b * CI_ + lane)) * NC_;       // prefetch rows
  unsigned* pw = &p_lds[wave][0];

  for (int it = 0; it < NC_ / 64; ++it) {
    const int k0 = it * 64;

    // K B-operands: lane l16 = key column; 16 contiguous dims per half-wave
    v16bf kb0 = *(const v16bf*)(Kbase + (size_t)(k0      + l16) * CI_ + half * 16);
    v16bf kb1 = *(const v16bf*)(Kbase + (size_t)(k0 + 16 + l16) * CI_ + half * 16);
    v16bf kb2 = *(const v16bf*)(Kbase + (size_t)(k0 + 32 + l16) * CI_ + half * 16);
    v16bf kb3 = *(const v16bf*)(Kbase + (size_t)(k0 + 48 + l16) * CI_ + half * 16);

    // prefetch next chunk: 64 K rows (2 x 32 lanes) + 32 V dim-rows
    __builtin_prefetch(Kpref + (size_t)(k0 + 64) * CI_, 0, 3);
    __builtin_prefetch(Kpref + (size_t)(k0 + 96) * CI_, 0, 3);
    __builtin_prefetch(Vpref + (k0 + 64), 0, 3);

    v8f s0 = {}, s1 = {}, s2 = {}, s3 = {};
    s0 = wmma_bf16(qa, kb0, s0);   // 16q x 16k, keys k0    .. k0+15
    s1 = wmma_bf16(qa, kb1, s1);   // keys k0+16 .. k0+31
    s2 = wmma_bf16(qa, kb2, s2);   // keys k0+32 .. k0+47
    s3 = wmma_bf16(qa, kb3, s3);   // keys k0+48 .. k0+63

    // ---- online softmax (per query row m = v + 8*half), DPP reductions ----
    float scl[8];
#pragma unroll
    for (int v = 0; v < 8; ++v) {
      float lmax = fmaxf(fmaxf(s0[v], s1[v]), fmaxf(s2[v], s3[v]));
      float mnew = fmaxf(mrun[v], red16_max(lmax));
      scl[v]   = __expf(mrun[v] - mnew);
      float p0 = __expf(s0[v] - mnew);
      float p1 = __expf(s1[v] - mnew);
      float p2 = __expf(s2[v] - mnew);
      float p3 = __expf(s3[v] - mnew);
      float rs = red16_sum((p0 + p1) + (p2 + p3));
      lrun[v] = lrun[v] * scl[v] + rs;
      mrun[v] = mnew;
      // packed P stores: two ds_store_b32 per row (tiles 0 and 1)
      int row = (v + 8 * half) * 16 + l16;
      pw[row]       = pack_bf16(p0, p1);
      pw[256 + row] = pack_bf16(p2, p3);
    }
    asm volatile("s_wait_dscnt 0" ::: "memory");

    // ---- P as A-operands: contiguous 32B spans, even/odd deinterleave ----
    v16bf pa0, pa1;
    {
      const bf16* prow0 = (const bf16*)pw + l16 * 32 + half * 16;
      const bf16* prow1 = (const bf16*)(pw + 256) + l16 * 32 + half * 16;
      v16bf ld0 = *(const v16bf*)prow0;
      v16bf ld1 = *(const v16bf*)prow1;
#pragma unroll
      for (int e = 0; e < 8; ++e) {
        pa0[e] = ld0[2 * e]; pa0[8 + e] = ld0[2 * e + 1];
        pa1[e] = ld1[2 * e]; pa1[8 + e] = ld1[2 * e + 1];
      }
    }

    // ---- V B-operands (transposed layout -> contiguous key runs) ----
    v16bf vb_lo_a = __builtin_shufflevector(*(const v8bf*)(Vrow0 + k0 + 8 * half), *(const v8bf*)(Vrow0 + k0 + 16 + 8 * half), 0,1,2,3,4,5,6,7,8,9,10,11,12,13,14,15);
    v16bf vb_hi_a = __builtin_shufflevector(*(const v8bf*)(Vrow1 + k0 + 8 * half), *(const v8bf*)(Vrow1 + k0 + 16 + 8 * half), 0,1,2,3,4,5,6,7,8,9,10,11,12,13,14,15);
    v16bf vb_lo_b = __builtin_shufflevector(*(const v8bf*)(Vrow0 + k0 + 32 + 8 * half), *(const v8bf*)(Vrow0 + k0 + 48 + 8 * half), 0,1,2,3,4,5,6,7,8,9,10,11,12,13,14,15);
    v16bf vb_hi_b = __builtin_shufflevector(*(const v8bf*)(Vrow1 + k0 + 32 + 8 * half), *(const v8bf*)(Vrow1 + k0 + 48 + 8 * half), 0,1,2,3,4,5,6,7,8,9,10,11,12,13,14,15);

#pragma unroll
    for (int v = 0; v < 8; ++v) { acc_lo[v] *= scl[v]; acc_hi[v] *= scl[v]; }
    acc_lo = wmma_bf16(pa0, vb_lo_a, acc_lo);
    acc_hi = wmma_bf16(pa0, vb_hi_a, acc_hi);
    acc_lo = wmma_bf16(pa1, vb_lo_b, acc_lo);
    acc_hi = wmma_bf16(pa1, vb_hi_b, acc_hi);
  }

  // ---- epilogue: divide by l and store [B][Nc][32] f32 ----
#pragma unroll
  for (int v = 0; v < 8; ++v) {
    int m = v + 8 * half;
    float inv = 1.0f / lrun[v];
    size_t base = ((size_t)(b * NC_ + q0 + m)) * CI_;
    O[base + l16]      = acc_lo[v] * inv;
    O[base + 16 + l16] = acc_hi[v] * inv;
  }
}

// ---------------------------------------------------------------------------
// Kernel 4a: zero the BN reduction buffer (128 floats)
// ---------------------------------------------------------------------------
__global__ void init_sums_kernel(float* __restrict__ sums) {
  if (threadIdx.x < 128) sums[threadIdx.x] = 0.0f;
}

// ---------------------------------------------------------------------------
// Kernel 4b: y = W(attn_out) + b  (1x1 conv 32->64), plus per-channel
// sum / sum-of-squares for training-mode BatchNorm.  block = (o, b).
// ---------------------------------------------------------------------------
__global__ __launch_bounds__(256)
void conv_w_kernel(const float* __restrict__ attn, const float* __restrict__ w_w,
                   const float* __restrict__ w_b, float* __restrict__ y,
                   float* __restrict__ sums) {
  const int o = blockIdx.x;
  const int b = blockIdx.y;
  __shared__ float r1[256], r2[256];

  float wr[CI_];
#pragma unroll
  for (int ci = 0; ci < CI_; ++ci) wr[ci] = w_w[o * CI_ + ci];
  const float bias = w_b[o];

  float s = 0.0f, s2 = 0.0f;
  for (int n = threadIdx.x; n < NC_; n += 256) {
    const float* a = attn + ((size_t)(b * NC_ + n)) * CI_;
    float v = bias;
#pragma unroll
    for (int ci = 0; ci < CI_; ++ci) v += wr[ci] * a[ci];
    y[((size_t)(b * C_ + o)) * NC_ + n] = v;
    s += v; s2 += v * v;
  }
  r1[threadIdx.x] = s; r2[threadIdx.x] = s2;
  __syncthreads();
  for (int off = 128; off > 0; off >>= 1) {
    if (threadIdx.x < off) {
      r1[threadIdx.x] += r1[threadIdx.x + off];
      r2[threadIdx.x] += r2[threadIdx.x + off];
    }
    __syncthreads();
  }
  if (threadIdx.x == 0) {
    atomicAdd(&sums[o], r1[0]);
    atomicAdd(&sums[C_ + o], r2[0]);
  }
}

// ---------------------------------------------------------------------------
// Kernel 4c: fold BN statistics into per-channel scale/shift.
// ---------------------------------------------------------------------------
__global__ void bn_stats_kernel(const float* __restrict__ sums,
                                const float* __restrict__ gamma,
                                const float* __restrict__ beta,
                                float* __restrict__ stats) {
  const int o = threadIdx.x;
  if (o < C_) {
    const float inv_n = 1.0f / (float)(B_ * NC_);
    float mean = sums[o] * inv_n;
    float var  = sums[C_ + o] * inv_n - mean * mean;   // biased variance
    float sc   = gamma[o] * rsqrtf(var + BN_EPS);
    stats[o]      = sc;
    stats[C_ + o] = beta[o] - mean * sc;
  }
}

// ---------------------------------------------------------------------------
// Kernel 5: bilinear upsample 64->128 of bn(y) + residual add of main.
// (per-channel affine commutes with bilinear interpolation)
// ---------------------------------------------------------------------------
__global__ __launch_bounds__(256)
void final_kernel(const float* __restrict__ y, const float* __restrict__ stats,
                  const float* __restrict__ main_in, float* __restrict__ out) {
  const int idx = blockIdx.x * 256 + threadIdx.x;   // b*C*128*128 linear
  const int j = idx & (HM_ - 1);
  const int i = (idx >> 7) & (HM_ - 1);
  const int c = (idx >> 14) & (C_ - 1);
  const int b = idx >> 20;

  float si = fminf(fmaxf((i + 0.5f) * 0.5f - 0.5f, 0.0f), (float)(HC_ - 1));
  float sj = fminf(fmaxf((j + 0.5f) * 0.5f - 0.5f, 0.0f), (float)(HC_ - 1));
  int i0 = (int)floorf(si), j0 = (int)floorf(sj);
  int i1 = min(i0 + 1, HC_ - 1), j1 = min(j0 + 1, HC_ - 1);
  float wi = si - (float)i0, wj = sj - (float)j0;

  const float* yc = y + ((size_t)(b * C_ + c)) * NC_;
  float v00 = yc[i0 * HC_ + j0], v01 = yc[i0 * HC_ + j1];
  float v10 = yc[i1 * HC_ + j0], v11 = yc[i1 * HC_ + j1];
  float top = v00 + (v01 - v00) * wj;
  float bot = v10 + (v11 - v10) * wj;
  float val = top + (bot - top) * wi;

  out[idx] = val * stats[c] + stats[C_ + c] + main_in[idx];
}

// ---------------------------------------------------------------------------
// Launch
// ---------------------------------------------------------------------------
extern "C" void kernel_launch(void* const* d_in, const int* in_sizes, int n_in,
                              void* d_out, int out_size, void* d_ws, size_t ws_size,
                              hipStream_t stream) {
  const float* main_in = (const float*)d_in[0];
  const float* cross   = (const float*)d_in[1];
  const float* g_w     = (const float*)d_in[2];
  const float* g_b     = (const float*)d_in[3];
  const float* th_w    = (const float*)d_in[4];
  const float* th_b    = (const float*)d_in[5];
  const float* ph_w    = (const float*)d_in[6];
  const float* ph_b    = (const float*)d_in[7];
  const float* w_w     = (const float*)d_in[8];
  const float* w_b     = (const float*)d_in[9];
  const float* gamma   = (const float*)d_in[10];
  const float* beta    = (const float*)d_in[11];
  float* out = (float*)d_out;

  // workspace carve-up (all offsets 256B aligned)
  char* ws = (char*)d_ws;
  bf16*  Qb    = (bf16*)(ws);                                // 1 MiB
  bf16*  Kb    = (bf16*)(ws + 1 * 1048576);                  // 1 MiB
  bf16*  Vt    = (bf16*)(ws + 2 * 1048576);                  // 1 MiB (transposed)
  float* attn  = (float*)(ws + 3 * 1048576);                 // 2 MiB
  float* yb    = (float*)(ws + 5 * 1048576);                 // 4 MiB
  float* sums  = (float*)(ws + 9 * 1048576);                 // 512 B
  float* stats = (float*)(ws + 9 * 1048576 + 1024);          // 512 B
  (void)ws_size; (void)in_sizes; (void)n_in; (void)out_size;

  const int bn = (B_ * NC_) / 256;   // 64 blocks for prep kernels

  prep_qk_kernel<<<bn, 256, 0, stream>>>(cross, g_w, g_b, th_w, th_b, Qb, Kb);
  prep_v_kernel<<<bn, 256, 0, stream>>>(main_in, ph_w, ph_b, Vt);
  flash_attn_kernel<<<dim3(NC_ / 128, B_), 256, 0, stream>>>(Qb, Kb, Vt, attn);
  init_sums_kernel<<<1, 128, 0, stream>>>(sums);
  conv_w_kernel<<<dim3(C_, B_), 256, 0, stream>>>(attn, w_w, w_b, yb, sums);
  bn_stats_kernel<<<1, 64, 0, stream>>>(sums, gamma, beta, stats);
  final_kernel<<<(B_ * C_ * HM_ * HM_) / 256, 256, 0, stream>>>(yb, stats, main_in, out);
}
